// minerva_old_35124242547423
// MI455X (gfx1250) — matrix operand order repaired and checked by reading, not compiled, not run on hardware
//
#include <hip/hip_runtime.h>

typedef __attribute__((ext_vector_type(16))) __bf16 v16bf;
typedef __attribute__((ext_vector_type(8)))  __bf16 v8bf;
typedef __attribute__((ext_vector_type(8)))  float  v8f;

#define WMMA_BF16 __builtin_amdgcn_wmma_f32_16x16x32_bf16

constexpr int Bn = 4096, Dd = 1024, Ff = 1024, Nn = 8192, Cc = 64, Ll = 28;
constexpr int ROWS = Bn + Nn;          // 12288 combined projection rows
constexpr int NSPLIT = 8;              // N-range splits per row-block
constexpr float FEPS = 1e-12f;

// ---------- WMMA fragment loaders (bf16, 16x16x32) ----------
// A 16x32: lane<16 -> M=lane, K = baseK(0)+0..7 and 16+baseK+0..7
//          lane>=16 -> M=lane-16, baseK=8
__device__ inline v16bf load_a_frag(const __bf16* base, int ld, int row0, int k0, int lane) {
  int m = lane & 15;
  int baseK = (lane < 16) ? 0 : 8;
  const __bf16* p = base + (size_t)(row0 + m) * ld + k0 + baseK;
  v8bf lo = *(const v8bf*)p;
  v8bf hi = *(const v8bf*)(p + 16);
  v16bf a;
#pragma unroll
  for (int i = 0; i < 8; ++i) { a[i] = lo[i]; a[i + 8] = hi[i]; }
  return a;
}

// B 32x16 (B[k][n] = M[n][k], M row-major with ld): lane<16 -> N=lane, K=0..7,16..23; lane>=16 -> K=8..15,24..31
__device__ inline v16bf load_b_frag(const __bf16* base, int ld, int col0, int k0, int lane) {
  int n = lane & 15;
  int baseK = (lane < 16) ? 0 : 8;
  const __bf16* p = base + (size_t)(col0 + n) * ld + k0 + baseK;
  v8bf lo = *(const v8bf*)p;
  v8bf hi = *(const v8bf*)(p + 16);
  v16bf b;
#pragma unroll
  for (int i = 0; i < 8; ++i) { b[i] = lo[i]; b[i + 8] = hi[i]; }
  return b;
}

// ---------- 1) fp32 -> bf16 converts ----------
__global__ void k_convert(const float* feat, const float* exf, const float* gw,
                          __bf16* Xbf, __bf16* Wbf) {
  size_t stride = (size_t)gridDim.x * blockDim.x;
  size_t i0 = (size_t)blockIdx.x * blockDim.x + threadIdx.x;
  const size_t totalX = (size_t)ROWS * Dd;
  const size_t nFeat  = (size_t)Bn * Dd;
  for (size_t i = i0; i < totalX; i += stride)
    Xbf[i] = (__bf16)((i < nFeat) ? feat[i] : exf[i - nFeat]);
  const size_t totalW = (size_t)Ff * Dd;
  for (size_t i = i0; i < totalW; i += stride)
    Wbf[i] = (__bf16)gw[i];
}

// ---------- 2) projection GEMM: Pout[12288,1024] = Xbf @ Wbf^T ----------
// Software-pipelined: next k-step's fragments are loaded while current WMMAs run.
__global__ void __launch_bounds__(128) k_proj(const __bf16* Xbf, const __bf16* Wbf, float* Pout) {
  int lane = threadIdx.x & 31, wv = threadIdx.x >> 5;
  int tile = blockIdx.x * 4 + wv;       // (ROWS/16)*(Ff/64) = 12288 wave-tiles
  int mt = tile >> 4, nt = tile & 15;
  int row0 = mt * 16, col0 = nt * 64;
  v8f acc[4] = {};

  v16bf a = load_a_frag(Xbf, Dd, row0, 0, lane);
  v16bf b0 = load_b_frag(Wbf, Dd, col0,      0, lane);
  v16bf b1 = load_b_frag(Wbf, Dd, col0 + 16, 0, lane);
  v16bf b2 = load_b_frag(Wbf, Dd, col0 + 32, 0, lane);
  v16bf b3 = load_b_frag(Wbf, Dd, col0 + 48, 0, lane);

  for (int k0 = 0; k0 < Dd - 32; k0 += 32) {
    int kn = k0 + 32;
    v16bf an  = load_a_frag(Xbf, Dd, row0, kn, lane);
    v16bf bn0 = load_b_frag(Wbf, Dd, col0,      kn, lane);
    v16bf bn1 = load_b_frag(Wbf, Dd, col0 + 16, kn, lane);
    v16bf bn2 = load_b_frag(Wbf, Dd, col0 + 32, kn, lane);
    v16bf bn3 = load_b_frag(Wbf, Dd, col0 + 48, kn, lane);
    acc[0] = WMMA_BF16(false, a, false, b0, (short)0, acc[0], false, false);
    acc[1] = WMMA_BF16(false, a, false, b1, (short)0, acc[1], false, false);
    acc[2] = WMMA_BF16(false, a, false, b2, (short)0, acc[2], false, false);
    acc[3] = WMMA_BF16(false, a, false, b3, (short)0, acc[3], false, false);
    a = an; b0 = bn0; b1 = bn1; b2 = bn2; b3 = bn3;
  }
  acc[0] = WMMA_BF16(false, a, false, b0, (short)0, acc[0], false, false);
  acc[1] = WMMA_BF16(false, a, false, b1, (short)0, acc[1], false, false);
  acc[2] = WMMA_BF16(false, a, false, b2, (short)0, acc[2], false, false);
  acc[3] = WMMA_BF16(false, a, false, b3, (short)0, acc[3], false, false);

  int hi = lane >> 4, n = lane & 15;
#pragma unroll
  for (int t = 0; t < 4; ++t)
#pragma unroll
    for (int v = 0; v < 8; ++v)
      Pout[(size_t)(row0 + v + 8 * hi) * Ff + col0 + 16 * t + n] = acc[t][v];
}

// ---------- 3) row L2 normalize -> bf16 ----------
__global__ void k_norm(const float* Pout, __bf16* XN) {
  int row = blockIdx.x;
  __shared__ float red[256];
  const float* p = Pout + (size_t)row * Ff;
  float s = 0.f;
  for (int c = threadIdx.x; c < Ff; c += 256) { float v = p[c]; s += v * v; }
  red[threadIdx.x] = s; __syncthreads();
  for (int off = 128; off > 0; off >>= 1) {
    if ((int)threadIdx.x < off) red[threadIdx.x] += red[threadIdx.x + off];
    __syncthreads();
  }
  float inv = 1.0f / fmaxf(sqrtf(red[0]), FEPS);
  for (int c = threadIdx.x; c < Ff; c += 256)
    XN[(size_t)row * Ff + c] = (__bf16)(p[c] * inv);
}

// ---------- 4) transpose ex_class_reps -> Et[64][N] bf16 ----------
__global__ void k_transE(const float* E, __bf16* Et) {
  int idx = blockIdx.x * 256 + threadIdx.x;
  if (idx < Nn * Cc) {
    int nrow = idx / Cc, c = idx % Cc;
    Et[(size_t)c * Nn + nrow] = (__bf16)E[idx];
  }
}

// ---------- 5) fused main: s=fn@exn^T, a=sign(s)|s|^p, rowsum, echo=a@E ----------
__global__ void __launch_bounds__(128) k_main(const __bf16* XN, const __bf16* Et, const int* pptr,
                                              float* echo_raw, float* rowsum) {
  constexpr int LDA = Dd + 8;
  __shared__ __bf16 Afull[16 * LDA];           // fn row-block, staged once
  __shared__ __bf16 a_lds[4][16][64];          // per-wave cubed tile (D->A relayout)
  __shared__ float echo_lds[16 * 64];
  __shared__ float rs_lds[16];

  int tid = threadIdx.x, lane = tid & 31, wv = tid >> 5;
  int rb = blockIdx.x / NSPLIT, ns = blockIdx.x % NSPLIT;
  int row0 = rb * 16;
  int pf = pptr[0];

  for (int i = tid; i < 16 * 64; i += 128) echo_lds[i] = 0.f;
  if (tid < 16) rs_lds[tid] = 0.f;
  for (int i = tid; i < 16 * (Dd / 8); i += 128) {
    int r = i / (Dd / 8), kc = (i % (Dd / 8)) * 8;
    *(v8bf*)&Afull[r * LDA + kc] = *(const v8bf*)&XN[(size_t)(row0 + r) * Dd + kc];
  }
  __syncthreads();

  int hi = lane >> 4, nlane = lane & 15;
  const __bf16* exn = XN + (size_t)Bn * Ff;    // normalized exemplar rows
  v8f eacc[4] = {};
  float rs[8] = {0.f, 0.f, 0.f, 0.f, 0.f, 0.f, 0.f, 0.f};

  for (int ch = wv; ch < 16; ch += 4) {        // 16 chunks of 64 N-cols per WG
    int nc0 = ns * 1024 + ch * 64;

    // warm L2/WGP$ for the next chunk this wave will process (global_prefetch_b8)
    if (ch + 4 < 16) {
      const __bf16* nxt = exn + (size_t)(nc0 + 256) * Ff;
      __builtin_prefetch(nxt + (size_t)lane * Ff, 0, 3);
      __builtin_prefetch(nxt + (size_t)(lane + 32) * Ff, 0, 3);
    }

    // ---- s-tile: software-pipelined K loop, A from LDS, B from global ----
    v8f s[4] = {};
    v16bf b0 = load_b_frag(exn, Ff, nc0,      0, lane);
    v16bf b1 = load_b_frag(exn, Ff, nc0 + 16, 0, lane);
    v16bf b2 = load_b_frag(exn, Ff, nc0 + 32, 0, lane);
    v16bf b3 = load_b_frag(exn, Ff, nc0 + 48, 0, lane);
    for (int k0 = 0; k0 < Ff - 32; k0 += 32) {
      int kn = k0 + 32;
      v16bf bn0 = load_b_frag(exn, Ff, nc0,      kn, lane);
      v16bf bn1 = load_b_frag(exn, Ff, nc0 + 16, kn, lane);
      v16bf bn2 = load_b_frag(exn, Ff, nc0 + 32, kn, lane);
      v16bf bn3 = load_b_frag(exn, Ff, nc0 + 48, kn, lane);
      v16bf a = load_a_frag(Afull, LDA, 0, k0, lane);
      s[0] = WMMA_BF16(false, a, false, b0, (short)0, s[0], false, false);
      s[1] = WMMA_BF16(false, a, false, b1, (short)0, s[1], false, false);
      s[2] = WMMA_BF16(false, a, false, b2, (short)0, s[2], false, false);
      s[3] = WMMA_BF16(false, a, false, b3, (short)0, s[3], false, false);
      b0 = bn0; b1 = bn1; b2 = bn2; b3 = bn3;
    }
    {
      v16bf a = load_a_frag(Afull, LDA, 0, Ff - 32, lane);
      s[0] = WMMA_BF16(false, a, false, b0, (short)0, s[0], false, false);
      s[1] = WMMA_BF16(false, a, false, b1, (short)0, s[1], false, false);
      s[2] = WMMA_BF16(false, a, false, b2, (short)0, s[2], false, false);
      s[3] = WMMA_BF16(false, a, false, b3, (short)0, s[3], false, false);
    }

    // power-sign activation + |a| row partials + stash bf16 tile for echo WMMA
#pragma unroll
    for (int t = 0; t < 4; ++t)
#pragma unroll
      for (int v = 0; v < 8; ++v) {
        float sv = s[t][v];
        float av = (pf == 3) ? sv * sv * sv
                             : copysignf(powf(fabsf(sv), (float)pf), sv);
        rs[v] += fabsf(av);
        a_lds[wv][v + 8 * hi][16 * t + nlane] = (__bf16)av;
      }
    // echo += a_tile[16x64] @ E_chunk[64x64]  (two K=32 steps, 4 C-tiles)
#pragma unroll
    for (int kk = 0; kk < 64; kk += 32) {
      v16bf a2 = load_a_frag(&a_lds[wv][0][0], 64, 0, kk, lane);
#pragma unroll
      for (int t = 0; t < 4; ++t) {
        v16bf e = load_b_frag(Et, Nn, 16 * t, nc0 + kk, lane);
        eacc[t] = WMMA_BF16(false, a2, false, e, (short)0, eacc[t], false, false);
      }
    }
  }

#pragma unroll
  for (int v = 0; v < 8; ++v) atomicAdd(&rs_lds[v + 8 * hi], rs[v]);
#pragma unroll
  for (int t = 0; t < 4; ++t)
#pragma unroll
    for (int v = 0; v < 8; ++v)
      atomicAdd(&echo_lds[(v + 8 * hi) * 64 + 16 * t + nlane], eacc[t][v]);
  __syncthreads();
  for (int i = tid; i < 16 * 64; i += 128)
    atomicAdd(&echo_raw[(size_t)(row0 + i / 64) * 64 + (i % 64)], echo_lds[i]);
  if (tid < 16) atomicAdd(&rowsum[row0 + tid], rs_lds[tid]);
}

// ---------- 6) finalize: scale echo, cdist, BCE ----------
__global__ void k_final(const float* echo_raw, const float* rowsum,
                        const float* class_reps, const float* labels,
                        float* out, float* lossAcc) {
  int b = blockIdx.x;
  __shared__ float echo_s[64];
  __shared__ float terms[Ll];
  int tid = threadIdx.x;   // 64 threads
  float inv = 1.0f / fmaxf(rowsum[b], FEPS);
  echo_s[tid] = echo_raw[(size_t)b * 64 + tid] * inv;
  __syncthreads();
  if (tid < Ll) {
    float d2 = 0.f;
    const float* cr = class_reps + tid * 64;
#pragma unroll 8
    for (int c = 0; c < 64; ++c) { float df = echo_s[c] - cr[c]; d2 += df * df; }
    float x = -sqrtf(d2);
    out[1 + (size_t)b * Ll + tid] = x;
    float y = labels[(size_t)b * Ll + tid];
    terms[tid] = fmaxf(x, 0.f) - x * y + log1pf(expf(-fabsf(x)));
  }
  __syncthreads();
  if (tid == 0) {
    float s = 0.f;
    for (int l = 0; l < Ll; ++l) s += terms[l];
    atomicAdd(lossAcc, s);
  }
}

__global__ void k_loss(const float* lossAcc, float* out) {
  out[0] = lossAcc[0] / (float)(Bn * Ll);
}

extern "C" void kernel_launch(void* const* d_in, const int* in_sizes, int n_in,
                              void* d_out, int out_size, void* d_ws, size_t ws_size,
                              hipStream_t stream) {
  const float* features      = (const float*)d_in[0];
  const float* labels        = (const float*)d_in[1];
  const float* g_weight      = (const float*)d_in[2];
  const float* ex_features   = (const float*)d_in[3];
  const float* ex_class_reps = (const float*)d_in[4];
  const float* class_reps    = (const float*)d_in[5];
  const int*   p_factor      = (const int*)d_in[6];
  float* out = (float*)d_out;

  char* ws = (char*)d_ws;
  size_t off = 0;
  auto alloc = [&](size_t bytes) -> void* {
    void* p = ws + off;
    off += (bytes + 255) & ~(size_t)255;
    return p;
  };
  __bf16* Xbf      = (__bf16*)alloc((size_t)ROWS * Dd * 2);  // reused as XN after k_norm
  __bf16* Wbf      = (__bf16*)alloc((size_t)Ff * Dd * 2);
  float*  Pout     = (float*)alloc((size_t)ROWS * Ff * 4);
  __bf16* Et       = (__bf16*)alloc((size_t)Cc * Nn * 2);
  float*  echo_raw = (float*)alloc((size_t)Bn * Cc * 4);
  float*  rowsum   = (float*)alloc((size_t)Bn * 4);
  float*  lossAcc  = (float*)alloc(256);
  __bf16* XN = Xbf;  // projection inputs dead after k_proj; reuse buffer

  hipMemsetAsync(echo_raw, 0, (size_t)Bn * Cc * 4, stream);
  hipMemsetAsync(rowsum,   0, (size_t)Bn * 4, stream);
  hipMemsetAsync(lossAcc,  0, 4, stream);

  k_convert<<<4096, 256, 0, stream>>>(features, ex_features, g_weight, Xbf, Wbf);
  k_proj<<<(ROWS / 16) * (Ff / 64) / 4, 128, 0, stream>>>(Xbf, Wbf, Pout);
  k_norm<<<ROWS, 256, 0, stream>>>(Pout, XN);
  k_transE<<<(Nn * Cc + 255) / 256, 256, 0, stream>>>(ex_class_reps, Et);
  k_main<<<(Bn / 16) * NSPLIT, 128, 0, stream>>>(XN, Et, p_factor, echo_raw, rowsum);
  k_final<<<Bn, 64, 0, stream>>>(echo_raw, rowsum, class_reps, labels, out, lossAcc);
  k_loss<<<1, 1, 0, stream>>>(lossAcc, out);
}
